// PointBertEncoder_23699629540069
// MI455X (gfx1250) — matrix operand order repaired and checked by reading, not compiled, not run on hardware
//
#include <hip/hip_runtime.h>
#include <hip/hip_bf16.h>
#include <math.h>

typedef __attribute__((ext_vector_type(16))) _Float16 v16h;
typedef __attribute__((ext_vector_type(4)))  _Float16 v4h;
typedef __attribute__((ext_vector_type(8)))  float    v8f;
typedef __attribute__((ext_vector_type(4)))  int      i4v;

#if defined(__has_builtin)
#if __has_builtin(__builtin_amdgcn_global_load_async_to_lds_b128) && \
    __has_builtin(__builtin_amdgcn_s_wait_asynccnt)
#define GEMM_ASYNC 1
#endif
#endif
#ifndef GEMM_ASYNC
#define GEMM_ASYNC 0
#endif

#if GEMM_ASYNC
typedef __attribute__((address_space(3))) i4v lds_i4v;
#endif

#define BB     16
#define NPTS   10000
#define CINCH  6
#define MCTR   512
#define KNN    32
#define EDIM   768
#define NHEAD  12
#define HDIM   64
#define NLAYER 12
#define MLPD   3072
#define SEQ    513
#define ROWS   (BB*SEQ)     /* 8208 */
#define LN_EPS 1e-5f

__device__ __forceinline__ float gelu_f(float v) {
    return 0.5f * v * (1.0f + erff(v * 0.70710678118654752f));
}

// ---------------------------------------------------------------- xyz extract
__global__ void k_extract_xyz(const float* __restrict__ pts, float* __restrict__ xyz) {
    int i = blockIdx.x * blockDim.x + threadIdx.x;
    if (i < BB * NPTS) {
        xyz[i * 3 + 0] = pts[i * CINCH + 0];
        xyz[i * 3 + 1] = pts[i * CINCH + 1];
        xyz[i * 3 + 2] = pts[i * CINCH + 2];
    }
}

// ---------------------------------------------------------------- FPS
__global__ __launch_bounds__(256) void k_fps(const float* __restrict__ xyz,
                                             float* __restrict__ dist,
                                             int* __restrict__ cidx) {
    int b = blockIdx.x;
    const float* X = xyz + (size_t)b * NPTS * 3;
    float* D = dist + (size_t)b * NPTS;
    __shared__ float rv[256];
    __shared__ int   ri[256];
    __shared__ int   farS;
    __shared__ float cx, cy, cz;
    int t = threadIdx.x;
    for (int i = t; i < NPTS; i += 256) D[i] = 1e10f;
    if (t == 0) farS = 0;
    __syncthreads();
    for (int s = 0; s < MCTR; s++) {
        if (t == 0) {
            int far = farS;
            cidx[b * MCTR + s] = far;
            cx = X[far * 3 + 0]; cy = X[far * 3 + 1]; cz = X[far * 3 + 2];
        }
        __syncthreads();
        float bv = -1.0f; int bi = 0;
        for (int i = t; i < NPTS; i += 256) {
            float dx = X[i * 3 + 0] - cx;
            float dy = X[i * 3 + 1] - cy;
            float dz = X[i * 3 + 2] - cz;
            float d  = dx * dx + dy * dy + dz * dz;
            float dn = fminf(D[i], d);
            D[i] = dn;
            if (dn > bv) { bv = dn; bi = i; }
        }
        rv[t] = bv; ri[t] = bi;
        __syncthreads();
        for (int ofs = 128; ofs > 0; ofs >>= 1) {
            if (t < ofs) {
                if (rv[t + ofs] > rv[t] || (rv[t + ofs] == rv[t] && ri[t + ofs] < ri[t])) {
                    rv[t] = rv[t + ofs]; ri[t] = ri[t + ofs];
                }
            }
            __syncthreads();
        }
        if (t == 0) farS = ri[0];
        __syncthreads();
    }
}

// ---------------------------------------------------------------- centers
__global__ void k_centers(const float* __restrict__ xyz, const int* __restrict__ cidx,
                          float* __restrict__ centers) {
    int i = blockIdx.x * blockDim.x + threadIdx.x;
    if (i < BB * MCTR) {
        int b = i / MCTR;
        int idx = cidx[i];
        centers[i * 3 + 0] = xyz[((size_t)b * NPTS + idx) * 3 + 0];
        centers[i * 3 + 1] = xyz[((size_t)b * NPTS + idx) * 3 + 1];
        centers[i * 3 + 2] = xyz[((size_t)b * NPTS + idx) * 3 + 2];
    }
}

// ---------------------------------------------------------------- kNN
__global__ __launch_bounds__(256) void k_knn(const float* __restrict__ xyz,
                                             const float* __restrict__ centers,
                                             int* __restrict__ gi) {
    int g = blockIdx.x;
    int b = g / MCTR;
    __shared__ float d2[NPTS];          // 40 KB in 320 KB LDS
    __shared__ float rv[256];
    __shared__ int   ri[256];
    const float* X = xyz + (size_t)b * NPTS * 3;
    float cx = centers[g * 3 + 0], cy = centers[g * 3 + 1], cz = centers[g * 3 + 2];
    int t = threadIdx.x;
    for (int i = t; i < NPTS; i += 256) {
        float dx = X[i * 3 + 0] - cx;
        float dy = X[i * 3 + 1] - cy;
        float dz = X[i * 3 + 2] - cz;
        d2[i] = dx * dx + dy * dy + dz * dz;
    }
    __syncthreads();
    for (int k = 0; k < KNN; k++) {
        float bv = 3e38f; int bi = 0;
        for (int i = t; i < NPTS; i += 256) {
            float v = d2[i];
            if (v < bv) { bv = v; bi = i; }
        }
        rv[t] = bv; ri[t] = bi;
        __syncthreads();
        for (int ofs = 128; ofs > 0; ofs >>= 1) {
            if (t < ofs) {
                if (rv[t + ofs] < rv[t] || (rv[t + ofs] == rv[t] && ri[t + ofs] < ri[t])) {
                    rv[t] = rv[t + ofs]; ri[t] = ri[t + ofs];
                }
            }
            __syncthreads();
        }
        if (t == 0) { gi[(size_t)g * KNN + k] = ri[0]; d2[ri[0]] = 3e38f; }
        __syncthreads();
    }
}

// ---------------------------------------------------------------- fused PointNet per group (layer 4 on WMMA)
__global__ __launch_bounds__(256) void k_pointnet(
    const float* __restrict__ pts, const float* __restrict__ xyz,
    const float* __restrict__ centers, const int* __restrict__ gi,
    const float* __restrict__ w1, const float* __restrict__ b1,
    const float* __restrict__ g1, const float* __restrict__ be1,
    const float* __restrict__ m1, const float* __restrict__ vv1,
    const float* __restrict__ w2, const float* __restrict__ b2,
    const float* __restrict__ g2, const float* __restrict__ be2,
    const float* __restrict__ m2, const float* __restrict__ vv2,
    const float* __restrict__ w3, const float* __restrict__ b3,
    const float* __restrict__ g3, const float* __restrict__ be3,
    const float* __restrict__ m3, const float* __restrict__ vv3,
    const float* __restrict__ w4, const float* __restrict__ b4,
    float* __restrict__ x) {
    int g = blockIdx.x;
    int b = g / MCTR, m = g % MCTR;
    __shared__ float    loc[KNN][9];
    __shared__ float    h1[KNN][64];
    __shared__ float    h2[KNN][128];
    __shared__ _Float16 h3h[KNN][264];   // f16 for WMMA, padded row stride
    int t = threadIdx.x;
    for (int i = t; i < KNN * 9; i += 256) {
        int p = i / 9, c = i % 9;
        int n = gi[(size_t)g * KNN + p];
        float v;
        if (c < 3) v = xyz[((size_t)b * NPTS + n) * 3 + c] - centers[g * 3 + c];
        else       v = pts[((size_t)b * NPTS + n) * CINCH + (c - 3)];
        loc[p][c] = v;
    }
    __syncthreads();
    for (int i = t; i < KNN * 64; i += 256) {
        int p = i >> 6, o = i & 63;
        float s = b1[o];
        #pragma unroll
        for (int j = 0; j < 9; j++) s += loc[p][j] * w1[o * 9 + j];
        s = (s - m1[o]) * rsqrtf(vv1[o] + LN_EPS) * g1[o] + be1[o];
        h1[p][o] = fmaxf(s, 0.0f);
    }
    __syncthreads();
    for (int i = t; i < KNN * 128; i += 256) {
        int p = i >> 7, o = i & 127;
        float s = b2[o];
        #pragma unroll 8
        for (int j = 0; j < 64; j++) s += h1[p][j] * w2[o * 64 + j];
        s = (s - m2[o]) * rsqrtf(vv2[o] + LN_EPS) * g2[o] + be2[o];
        h2[p][o] = fmaxf(s, 0.0f);
    }
    __syncthreads();
    for (int i = t; i < KNN * 256; i += 256) {
        int p = i >> 8, o = i & 255;
        float s = b3[o];
        #pragma unroll 8
        for (int j = 0; j < 128; j++) s += h2[p][j] * w3[o * 128 + j];
        s = (s - m3[o]) * rsqrtf(vv3[o] + LN_EPS) * g3[o] + be3[o];
        h3h[p][o] = (_Float16)fmaxf(s, 0.0f);
    }
    __syncthreads();
    // layer 4 via WMMA: (32 x 256) @ W4^T -> (32 x 768), then max-pool over the 32 points.
    int w = t >> 5, lane = t & 31, lr = lane & 15, hs = lane >> 4;
    for (int nt = w; nt < EDIM / 16; nt += 8) {      // 48 column tiles over 8 waves
        v8f a0 = {0.f,0.f,0.f,0.f,0.f,0.f,0.f,0.f};
        v8f a1 = {0.f,0.f,0.f,0.f,0.f,0.f,0.f,0.f};
        int e = nt * 16 + lr;
        #pragma unroll
        for (int ks = 0; ks < 8; ks++) {
            int kb = ks * 32;
            v16h af0, af1, bf;
            int k0 = hs ? 8 : 0;
            #pragma unroll
            for (int j = 0; j < 8; j++) {
                af0[j]     = h3h[lr][kb + k0 + j];
                af0[j + 8] = h3h[lr][kb + 16 + k0 + j];
                af1[j]     = h3h[16 + lr][kb + k0 + j];
                af1[j + 8] = h3h[16 + lr][kb + 16 + k0 + j];
            }
            int kbb = hs ? 8 : 0;
            #pragma unroll
            for (int j = 0; j < 16; j++)
                bf[j] = (_Float16)w4[(size_t)e * 256 + kb + kbb + j + (j & 8)];
            a0 = __builtin_amdgcn_wmma_f32_16x16x32_f16(false, af0, false, bf, (short)0, a0, false, false);
            a1 = __builtin_amdgcn_wmma_f32_16x16x32_f16(false, af1, false, bf, (short)0, a1, false, false);
        }
        float mx0 = -3e38f, mx1 = -3e38f;
        #pragma unroll
        for (int r = 0; r < 8; r++) { mx0 = fmaxf(mx0, a0[r]); mx1 = fmaxf(mx1, a1[r]); }
        mx0 = fmaxf(mx0, __shfl_xor(mx0, 16, 32));
        mx1 = fmaxf(mx1, __shfl_xor(mx1, 16, 32));
        float mx = fmaxf(mx0, mx1) + b4[e];
        if (hs == 0)
            x[((size_t)(b * SEQ + 1 + m)) * EDIM + e] = mx;
    }
}

// ---------------------------------------------------------------- pos embedding
__global__ __launch_bounds__(128) void k_pos(const float* __restrict__ centers,
                                             const float* __restrict__ pw1, const float* __restrict__ pb1,
                                             const float* __restrict__ pw2, const float* __restrict__ pb2,
                                             float* __restrict__ x) {
    int g = blockIdx.x;
    int b = g / MCTR, m = g % MCTR;
    __shared__ float hm[128];
    int t = threadIdx.x;
    if (t < 128) {
        float s = pb1[t];
        #pragma unroll
        for (int j = 0; j < 3; j++) s += centers[g * 3 + j] * pw1[t * 3 + j];
        hm[t] = gelu_f(s);
    }
    __syncthreads();
    for (int e = t; e < EDIM; e += 128) {
        float s = pb2[e];
        #pragma unroll 8
        for (int j = 0; j < 128; j++) s += hm[j] * pw2[(size_t)e * 128 + j];
        x[((size_t)(b * SEQ + 1 + m)) * EDIM + e] += s;
    }
}

// ---------------------------------------------------------------- cls token
__global__ void k_cls(const float* __restrict__ clst, const float* __restrict__ clsp,
                      float* __restrict__ x) {
    int b = blockIdx.x;
    for (int e = threadIdx.x; e < EDIM; e += blockDim.x)
        x[(size_t)(b * SEQ) * EDIM + e] = clst[e] + clsp[e];
}

// ---------------------------------------------------------------- LayerNorm
__global__ __launch_bounds__(256) void k_ln(const float* __restrict__ in,
                                            const float* __restrict__ gg,
                                            const float* __restrict__ bb,
                                            float* __restrict__ out, int rows) {
    int r = blockIdx.x;
    if (r >= rows) return;
    __shared__ float red[256];
    const float* xr = in + (size_t)r * EDIM;
    int t = threadIdx.x;
    float s = 0.0f;
    for (int i = t; i < EDIM; i += 256) s += xr[i];
    red[t] = s; __syncthreads();
    for (int ofs = 128; ofs > 0; ofs >>= 1) { if (t < ofs) red[t] += red[t + ofs]; __syncthreads(); }
    float mu = red[0] / (float)EDIM;
    __syncthreads();
    float s2 = 0.0f;
    for (int i = t; i < EDIM; i += 256) { float d = xr[i] - mu; s2 += d * d; }
    red[t] = s2; __syncthreads();
    for (int ofs = 128; ofs > 0; ofs >>= 1) { if (t < ofs) red[t] += red[t + ofs]; __syncthreads(); }
    float inv = rsqrtf(red[0] / (float)EDIM + LN_EPS);
    __syncthreads();
    for (int i = t; i < EDIM; i += 256)
        out[(size_t)r * EDIM + i] = (xr[i] - mu) * inv * gg[i] + bb[i];
}

// ---------------------------------------------------------------- WMMA GEMM: C = A @ W^T + bias (+gelu / +residual)
// A: [Mrows x Kd] f32.  W: [Nout x Kd] f32.  Block 256 thr = 8 waves (2Mx4N),
// wave tile 32x32 (2x2 WMMA), block tile 64x128, K-step 32, f16 in / f32 accum.
#if GEMM_ASYNC
// Double-buffered pipeline: GLOBAL_LOAD_ASYNC_TO_LDS_B128 stages tile k+1 (f32)
// while WMMAs consume tile k; settle with s_wait_asynccnt 0 + barrier.
// f32->f16 conversion happens at fragment build (VALU co-executes with WMMA).
__global__ __launch_bounds__(256) void k_gemm(const float* __restrict__ A,
                                              const float* __restrict__ W,
                                              const float* __restrict__ bias,
                                              const float* __restrict__ resid,
                                              float* __restrict__ C,
                                              int Mrows, int Nout, int Kd, int act) {
    __shared__ float Afl[2][64][32];     // 16 KB
    __shared__ float Wfl[2][128][32];    // 32 KB
    int t = threadIdx.x, w = t >> 5, lane = t & 31;
    int wr = w >> 2, wc = w & 3;
    int lr = lane & 15;
    int hs = lane >> 4;
    int m0 = blockIdx.x * 64, n0 = blockIdx.y * 128;
    v8f acc00 = {0.f,0.f,0.f,0.f,0.f,0.f,0.f,0.f};
    v8f acc01 = acc00, acc10 = acc00, acc11 = acc00;
    auto issue_tile = [&](int buf, int kb) {
        #pragma unroll
        for (int sIt = 0; sIt < 2; sIt++) {               // A: 512 b128 copies
            int s = t + sIt * 256;
            int r = s >> 3, kq = (s & 7) << 2;
            int row = m0 + r; row = (row < Mrows) ? row : (Mrows - 1);
            __builtin_amdgcn_global_load_async_to_lds_b128(
                (i4v*)&A[(size_t)row * Kd + kb + kq],
                (lds_i4v*)&Afl[buf][r][kq], 0, 0);
        }
        #pragma unroll
        for (int sIt = 0; sIt < 4; sIt++) {               // W: 1024 b128 copies
            int s = t + sIt * 256;
            int r = s >> 3, kq = (s & 7) << 2;
            __builtin_amdgcn_global_load_async_to_lds_b128(
                (i4v*)&W[(size_t)(n0 + r) * Kd + kb + kq],
                (lds_i4v*)&Wfl[buf][r][kq], 0, 0);
        }
    };
    issue_tile(0, 0);
    int buf = 0;
    for (int kb = 0; kb < Kd; kb += 32) {
        __builtin_amdgcn_s_wait_asynccnt(0);
        __syncthreads();          // tile `buf` ready; everyone done reading buf^1
        if (kb + 32 < Kd) issue_tile(buf ^ 1, kb + 32);
        v16h af0, af1, bf0, bf1;
        int aRow0 = wr * 32 + lr, aRow1 = aRow0 + 16;
        int k0 = hs ? 8 : 0;
        #pragma unroll
        for (int j = 0; j < 8; j++) {
            af0[j]     = (_Float16)Afl[buf][aRow0][k0 + j];
            af0[j + 8] = (_Float16)Afl[buf][aRow0][16 + k0 + j];
            af1[j]     = (_Float16)Afl[buf][aRow1][k0 + j];
            af1[j + 8] = (_Float16)Afl[buf][aRow1][16 + k0 + j];
        }
        int bCol0 = wc * 32 + lr, bCol1 = bCol0 + 16;
        int kb2 = hs ? 8 : 0;
        #pragma unroll
        for (int j = 0; j < 16; j++) {
            bf0[j] = (_Float16)Wfl[buf][bCol0][kb2 + j + (j & 8)];
            bf1[j] = (_Float16)Wfl[buf][bCol1][kb2 + j + (j & 8)];
        }
        acc00 = __builtin_amdgcn_wmma_f32_16x16x32_f16(false, af0, false, bf0, (short)0, acc00, false, false);
        acc01 = __builtin_amdgcn_wmma_f32_16x16x32_f16(false, af0, false, bf1, (short)0, acc01, false, false);
        acc10 = __builtin_amdgcn_wmma_f32_16x16x32_f16(false, af1, false, bf0, (short)0, acc10, false, false);
        acc11 = __builtin_amdgcn_wmma_f32_16x16x32_f16(false, af1, false, bf1, (short)0, acc11, false, false);
        buf ^= 1;
    }
    #pragma unroll
    for (int i = 0; i < 2; i++) {
        #pragma unroll
        for (int jj = 0; jj < 2; jj++) {
            v8f acc = (i == 0) ? (jj == 0 ? acc00 : acc01) : (jj == 0 ? acc10 : acc11);
            int n = n0 + wc * 32 + jj * 16 + lr;
            float bn = bias[n];
            #pragma unroll
            for (int r = 0; r < 8; r++) {
                int mm = m0 + wr * 32 + i * 16 + r + (hs ? 8 : 0);
                if (mm < Mrows) {
                    float v = acc[r] + bn;
                    if (act == 1) v = gelu_f(v);
                    if (resid) v += resid[(size_t)mm * Nout + n];
                    C[(size_t)mm * Nout + n] = v;
                }
            }
        }
    }
}
#else
// Synchronous staging fallback (float4 loads, packed v4h LDS stores).
__global__ __launch_bounds__(256) void k_gemm(const float* __restrict__ A,
                                              const float* __restrict__ W,
                                              const float* __restrict__ bias,
                                              const float* __restrict__ resid,
                                              float* __restrict__ C,
                                              int Mrows, int Nout, int Kd, int act) {
    __shared__ _Float16 As[64][36];
    __shared__ _Float16 Ws[128][36];
    int t = threadIdx.x, w = t >> 5, lane = t & 31;
    int wr = w >> 2, wc = w & 3;
    int lr = lane & 15;
    int hs = lane >> 4;
    int m0 = blockIdx.x * 64, n0 = blockIdx.y * 128;
    v8f acc00 = {0.f,0.f,0.f,0.f,0.f,0.f,0.f,0.f};
    v8f acc01 = acc00, acc10 = acc00, acc11 = acc00;
    for (int kb = 0; kb < Kd; kb += 32) {
        __syncthreads();
        #pragma unroll
        for (int sIt = 0; sIt < 2; sIt++) {
            int s = t + sIt * 256;
            int r = s >> 3, kq = (s & 7) << 2;
            int row = m0 + r; row = (row < Mrows) ? row : (Mrows - 1);
            float4 v = *(const float4*)&A[(size_t)row * Kd + kb + kq];
            v4h h; h[0] = (_Float16)v.x; h[1] = (_Float16)v.y;
                   h[2] = (_Float16)v.z; h[3] = (_Float16)v.w;
            *(v4h*)&As[r][kq] = h;
        }
        #pragma unroll
        for (int sIt = 0; sIt < 4; sIt++) {
            int s = t + sIt * 256;
            int r = s >> 3, kq = (s & 7) << 2;
            float4 v = *(const float4*)&W[(size_t)(n0 + r) * Kd + kb + kq];
            v4h h; h[0] = (_Float16)v.x; h[1] = (_Float16)v.y;
                   h[2] = (_Float16)v.z; h[3] = (_Float16)v.w;
            *(v4h*)&Ws[r][kq] = h;
        }
        if (kb + 32 < Kd) {
            __builtin_prefetch(&A[(size_t)(m0 + (t & 63)) * Kd + kb + 32], 0, 3);
            __builtin_prefetch(&W[(size_t)(n0 + (t & 127)) * Kd + kb + 32], 0, 3);
        }
        __syncthreads();
        v16h af0, af1, bf0, bf1;
        int aRow0 = wr * 32 + lr, aRow1 = aRow0 + 16;
        int k0 = hs ? 8 : 0;
        #pragma unroll
        for (int j = 0; j < 8; j++) {
            af0[j]     = As[aRow0][k0 + j];
            af0[j + 8] = As[aRow0][16 + k0 + j];
            af1[j]     = As[aRow1][k0 + j];
            af1[j + 8] = As[aRow1][16 + k0 + j];
        }
        int bCol0 = wc * 32 + lr, bCol1 = bCol0 + 16;
        int kb2 = hs ? 8 : 0;
        #pragma unroll
        for (int j = 0; j < 16; j++) {
            bf0[j] = Ws[bCol0][kb2 + j + (j & 8)];
            bf1[j] = Ws[bCol1][kb2 + j + (j & 8)];
        }
        acc00 = __builtin_amdgcn_wmma_f32_16x16x32_f16(false, af0, false, bf0, (short)0, acc00, false, false);
        acc01 = __builtin_amdgcn_wmma_f32_16x16x32_f16(false, af0, false, bf1, (short)0, acc01, false, false);
        acc10 = __builtin_amdgcn_wmma_f32_16x16x32_f16(false, af1, false, bf0, (short)0, acc10, false, false);
        acc11 = __builtin_amdgcn_wmma_f32_16x16x32_f16(false, af1, false, bf1, (short)0, acc11, false, false);
    }
    #pragma unroll
    for (int i = 0; i < 2; i++) {
        #pragma unroll
        for (int jj = 0; jj < 2; jj++) {
            v8f acc = (i == 0) ? (jj == 0 ? acc00 : acc01) : (jj == 0 ? acc10 : acc11);
            int n = n0 + wc * 32 + jj * 16 + lr;
            float bn = bias[n];
            #pragma unroll
            for (int r = 0; r < 8; r++) {
                int mm = m0 + wr * 32 + i * 16 + r + (hs ? 8 : 0);
                if (mm < Mrows) {
                    float v = acc[r] + bn;
                    if (act == 1) v = gelu_f(v);
                    if (resid) v += resid[(size_t)mm * Nout + n];
                    C[(size_t)mm * Nout + n] = v;
                }
            }
        }
    }
}
#endif

// ---------------------------------------------------------------- fused attention per (b, h, q-tile of 16)
__global__ __launch_bounds__(128) void k_attn(const float* __restrict__ qkv,
                                              float* __restrict__ obuf) {
    int bid = blockIdx.x;
    int qt = bid % 33;
    int h  = (bid / 33) % NHEAD;
    int b  = bid / (33 * NHEAD);
    __shared__ _Float16 Qs[16][64];
    __shared__ _Float16 KV[4][16][64];        // per-wave K slots; reused as Vs[64][32]
    __shared__ float    S[16][544];           // scores f32; probs f16 aliased in place
    __shared__ float    red[128];
    _Float16* P = (_Float16*)&S[0][0];
    int t = threadIdx.x, w = t >> 5, lane = t & 31, lr = lane & 15;
    int hs = lane >> 4;
    const float scale = 0.125f;               // 1/sqrt(64) folded into Q
    for (int i = t; i < 1024; i += 128) {
        int r = i >> 6, d = i & 63;
        int tok = qt * 16 + r;
        Qs[r][d] = (_Float16)((tok < SEQ)
                   ? qkv[(size_t)(b * SEQ + tok) * (3 * EDIM) + h * 64 + d] * scale : 0.0f);
    }
    for (int i = t; i < 256; i += 128) {
        int r = i >> 4, c = 528 + (i & 15);
        S[r][c] = -1e30f;
    }
    __syncthreads();
    v16h aq0, aq1;
    {
        int k0 = hs ? 8 : 0;
        #pragma unroll
        for (int j = 0; j < 8; j++) {
            aq0[j] = Qs[lr][k0 + j];       aq0[j + 8] = Qs[lr][16 + k0 + j];
            aq1[j] = Qs[lr][32 + k0 + j];  aq1[j + 8] = Qs[lr][48 + k0 + j];
        }
    }
    for (int kt = w; kt < 33; kt += 4) {
        for (int i = lane; i < 1024; i += 32) {
            int r = i >> 6, d = i & 63;
            int tok = kt * 16 + r;
            KV[w][r][d] = (_Float16)((tok < SEQ)
                          ? qkv[(size_t)(b * SEQ + tok) * (3 * EDIM) + EDIM + h * 64 + d] : 0.0f);
        }
        v16h b0, b1;
        int kb = hs ? 8 : 0;
        #pragma unroll
        for (int j = 0; j < 16; j++) {
            b0[j] = KV[w][lr][kb + j + (j & 8)];
            b1[j] = KV[w][lr][32 + kb + j + (j & 8)];
        }
        v8f c = {0.f,0.f,0.f,0.f,0.f,0.f,0.f,0.f};
        c = __builtin_amdgcn_wmma_f32_16x16x32_f16(false, aq0, false, b0, (short)0, c, false, false);
        c = __builtin_amdgcn_wmma_f32_16x16x32_f16(false, aq1, false, b1, (short)0, c, false, false);
        int key = kt * 16 + lr;
        #pragma unroll
        for (int r = 0; r < 8; r++) {
            int mm = r + (hs ? 8 : 0);
            S[mm][key] = (key < SEQ) ? c[r] : -1e30f;
        }
    }
    __syncthreads();
    for (int r = 0; r < 16; r++) {
        float lm = -3e38f;
        for (int c = t; c < 544; c += 128) lm = fmaxf(lm, S[r][c]);
        red[t] = lm; __syncthreads();
        for (int ofs = 64; ofs > 0; ofs >>= 1) { if (t < ofs) red[t] = fmaxf(red[t], red[t + ofs]); __syncthreads(); }
        float rmax = red[0]; __syncthreads();
        float ls = 0.0f;
        for (int c = t; c < 544; c += 128) { float e = __expf(S[r][c] - rmax); S[r][c] = e; ls += e; }
        red[t] = ls; __syncthreads();
        for (int ofs = 64; ofs > 0; ofs >>= 1) { if (t < ofs) red[t] += red[t + ofs]; __syncthreads(); }
        float inv = 1.0f / red[0]; __syncthreads();
        float tmp[5]; int cnt = 0;
        for (int c = t; c < 544; c += 128) tmp[cnt++] = S[r][c] * inv;
        __syncthreads();
        cnt = 0;
        for (int c = t; c < 544; c += 128) P[r * 544 + c] = (_Float16)tmp[cnt++];
        __syncthreads();
    }
    v8f co = {0.f,0.f,0.f,0.f,0.f,0.f,0.f,0.f};
    _Float16* Vs = &KV[0][0][0];              // reused as [64 dims][32 tokens]
    for (int ts = 0; ts < 17; ts++) {
        __syncthreads();
        for (int i = t; i < 2048; i += 128) {
            int d = i >> 5, j = i & 31;
            int tok = ts * 32 + j;
            Vs[d * 32 + j] = (_Float16)((tok < SEQ)
                             ? qkv[(size_t)(b * SEQ + tok) * (3 * EDIM) + 2 * EDIM + h * 64 + d] : 0.0f);
        }
        __syncthreads();
        v16h ap, bv;
        int k0 = hs ? 8 : 0;
        #pragma unroll
        for (int j = 0; j < 8; j++) {
            ap[j]     = P[lr * 544 + ts * 32 + k0 + j];
            ap[j + 8] = P[lr * 544 + ts * 32 + 16 + k0 + j];
        }
        int kb = hs ? 8 : 0;
        #pragma unroll
        for (int j = 0; j < 16; j++) bv[j] = Vs[(w * 16 + lr) * 32 + kb + j + (j & 8)];
        co = __builtin_amdgcn_wmma_f32_16x16x32_f16(false, ap, false, bv, (short)0, co, false, false);
    }
    int col = h * 64 + w * 16 + lr;
    #pragma unroll
    for (int r = 0; r < 8; r++) {
        int token = qt * 16 + r + (hs ? 8 : 0);
        if (token < SEQ)
            obuf[(size_t)(b * SEQ + token) * EDIM + col] = co[r];
    }
}

// ================================================================ launcher
extern "C" void kernel_launch(void* const* d_in, const int* in_sizes, int n_in,
                              void* d_out, int out_size, void* d_ws, size_t ws_size,
                              hipStream_t stream) {
    const float* pts    = (const float*)d_in[0];
    const float* pn_w1  = (const float*)d_in[1];
    const float* pn_b1  = (const float*)d_in[2];
    const float* bn1_g  = (const float*)d_in[3];
    const float* bn1_b  = (const float*)d_in[4];
    const float* bn1_m  = (const float*)d_in[5];
    const float* bn1_v  = (const float*)d_in[6];
    const float* pn_w2  = (const float*)d_in[7];
    const float* pn_b2  = (const float*)d_in[8];
    const float* bn2_g  = (const float*)d_in[9];
    const float* bn2_b  = (const float*)d_in[10];
    const float* bn2_m  = (const float*)d_in[11];
    const float* bn2_v  = (const float*)d_in[12];
    const float* pn_w3  = (const float*)d_in[13];
    const float* pn_b3  = (const float*)d_in[14];
    const float* bn3_g  = (const float*)d_in[15];
    const float* bn3_b  = (const float*)d_in[16];
    const float* bn3_m  = (const float*)d_in[17];
    const float* bn3_v  = (const float*)d_in[18];
    const float* pn_w4  = (const float*)d_in[19];
    const float* pn_b4  = (const float*)d_in[20];
    const float* pos_w1 = (const float*)d_in[21];
    const float* pos_b1 = (const float*)d_in[22];
    const float* pos_w2 = (const float*)d_in[23];
    const float* pos_b2 = (const float*)d_in[24];
    const float* cls_t  = (const float*)d_in[25];
    const float* cls_p  = (const float*)d_in[26];
    const float* ln1_g  = (const float*)d_in[27];
    const float* ln1_b  = (const float*)d_in[28];
    const float* qkv_w  = (const float*)d_in[29];
    const float* qkv_b  = (const float*)d_in[30];
    const float* out_w  = (const float*)d_in[31];
    const float* out_b  = (const float*)d_in[32];
    const float* ln2_g  = (const float*)d_in[33];
    const float* ln2_b  = (const float*)d_in[34];
    const float* fc1_w  = (const float*)d_in[35];
    const float* fc1_b  = (const float*)d_in[36];
    const float* fc2_w  = (const float*)d_in[37];
    const float* fc2_b  = (const float*)d_in[38];
    const float* norm_g = (const float*)d_in[39];
    const float* norm_b = (const float*)d_in[40];
    (void)in_sizes; (void)n_in; (void)out_size; (void)ws_size;

    char* ws = (char*)d_ws;
    size_t off = 0;
    auto alloc = [&](size_t bytes) -> void* {
        void* p = ws + off;
        off += (bytes + 255) & ~(size_t)255;
        return p;
    };
    float* xyz     = (float*)alloc((size_t)BB * NPTS * 3 * 4);
    float* dist    = (float*)alloc((size_t)BB * NPTS * 4);
    int*   cidx    = (int*)  alloc((size_t)BB * MCTR * 4);
    float* centers = (float*)alloc((size_t)BB * MCTR * 3 * 4);
    int*   gi      = (int*)  alloc((size_t)BB * MCTR * KNN * 4);
    float* x       = (float*)alloc((size_t)ROWS * EDIM * 4);
    float* hh      = (float*)alloc((size_t)ROWS * EDIM * 4);
    float* qkvb    = (float*)alloc((size_t)ROWS * 3 * EDIM * 4);
    float* ob      = (float*)alloc((size_t)ROWS * EDIM * 4);
    float* mlp     = (float*)alloc((size_t)ROWS * MLPD * 4);

    k_extract_xyz<<<(BB * NPTS + 255) / 256, 256, 0, stream>>>(pts, xyz);
    k_fps<<<BB, 256, 0, stream>>>(xyz, dist, cidx);
    k_centers<<<(BB * MCTR + 255) / 256, 256, 0, stream>>>(xyz, cidx, centers);
    k_knn<<<BB * MCTR, 256, 0, stream>>>(xyz, centers, gi);
    k_pointnet<<<BB * MCTR, 256, 0, stream>>>(pts, xyz, centers, gi,
        pn_w1, pn_b1, bn1_g, bn1_b, bn1_m, bn1_v,
        pn_w2, pn_b2, bn2_g, bn2_b, bn2_m, bn2_v,
        pn_w3, pn_b3, bn3_g, bn3_b, bn3_m, bn3_v,
        pn_w4, pn_b4, x);
    k_pos<<<BB * MCTR, 128, 0, stream>>>(centers, pos_w1, pos_b1, pos_w2, pos_b2, x);
    k_cls<<<BB, 256, 0, stream>>>(cls_t, cls_p, x);

    const int gM = (ROWS + 63) / 64;   // 129
    for (int l = 0; l < NLAYER; l++) {
        k_ln<<<ROWS, 256, 0, stream>>>(x, ln1_g + (size_t)l * EDIM, ln1_b + (size_t)l * EDIM, hh, ROWS);
        k_gemm<<<dim3(gM, (3 * EDIM) / 128), 256, 0, stream>>>(
            hh, qkv_w + (size_t)l * 3 * EDIM * EDIM, qkv_b + (size_t)l * 3 * EDIM,
            nullptr, qkvb, ROWS, 3 * EDIM, EDIM, 0);
        k_attn<<<BB * NHEAD * 33, 128, 0, stream>>>(qkvb, ob);
        k_gemm<<<dim3(gM, EDIM / 128), 256, 0, stream>>>(
            ob, out_w + (size_t)l * EDIM * EDIM, out_b + (size_t)l * EDIM,
            x, x, ROWS, EDIM, EDIM, 0);
        k_ln<<<ROWS, 256, 0, stream>>>(x, ln2_g + (size_t)l * EDIM, ln2_b + (size_t)l * EDIM, hh, ROWS);
        k_gemm<<<dim3(gM, MLPD / 128), 256, 0, stream>>>(
            hh, fc1_w + (size_t)l * MLPD * EDIM, fc1_b + (size_t)l * MLPD,
            nullptr, mlp, ROWS, MLPD, EDIM, 1);
        k_gemm<<<dim3(gM, EDIM / 128), 256, 0, stream>>>(
            mlp, fc2_w + (size_t)l * EDIM * MLPD, fc2_b + (size_t)l * EDIM,
            x, x, ROWS, EDIM, MLPD, 0);
    }
    k_ln<<<ROWS, 256, 0, stream>>>(x, norm_g, norm_b, (float*)d_out, ROWS);
}
